// PNA_4879082848576
// MI455X (gfx1250) — matrix operand order repaired and checked by reading, not compiled, not run on hardware
//
#include <hip/hip_runtime.h>

// ---------------------------------------------------------------------------
// PNA layer for MI455X (gfx1250, wave32, WMMA).
//
// Pipeline (all on one stream):
//   1) x -> bf16; pack W_pre/W_post/W_lin into WMMA B-fragment order (bf16)
//   2) CSR-bin edges by dst (count -> scan -> scatter src ids)
//   3) fused per-16-node-tile kernel:
//        hd = x[node] @ W_pre[0:128] + b_pre            (4 WMMAs / wave)
//        for each 32-edge group of the tile's CSR slice (double-buffered
//        global gather of x[src] overlapped with compute):
//            hs = x[src] @ W_pre[128:256]               (8 WMMAs / wave)
//            h  = hs + hd[local_node]
//            LDS-atomic sum / sum^2 / max / min per (node, col)
//        feat = [x | mean | max | min | std] * degree scaler   (bf16, LDS)
//        t    = feat @ W_post + b_post                  (20 WMMAs / wave)
//        out  = relu(t @ W_lin + b_lin)                 (4 WMMAs / wave)
//
// Rationale: the edge GEMM (52 GFLOP bf16) is ~25us of WMMA work; 410M global
// L2 atomics would be ~1ms. CSR + LDS atomics keeps the whole reduction
// on-chip; HBM traffic is ~50MB (~2us at 23.3 TB/s). x (bf16, 12.8MB) and the
// packed weights live in the 192MB L2, so the per-edge gather is L2-resident.
// ---------------------------------------------------------------------------

typedef __attribute__((ext_vector_type(16))) __bf16 bf16x16;
typedef __attribute__((ext_vector_type(8)))  float  f32x8;

union Frag {
    bf16x16 v;
    uint4   q[2];
    unsigned short s[16];
};

__device__ __forceinline__ unsigned short f2bf(float f) {
    unsigned u = __float_as_uint(f);
    u += 0x7fffu + ((u >> 16) & 1u);   // round-to-nearest-even
    return (unsigned short)(u >> 16);
}
// Order-preserving float -> u32 key (monotone under unsigned compare)
__device__ __forceinline__ unsigned flipf(float f) {
    unsigned u = __float_as_uint(f);
    return (u & 0x80000000u) ? ~u : (u | 0x80000000u);
}
__device__ __forceinline__ float unflipf(unsigned k) {
    unsigned u = (k & 0x80000000u) ? (k & 0x7fffffffu) : ~k;
    return __uint_as_float(u);
}

// --------------------------- prep kernels ----------------------------------

__global__ void k_convert_x(const float* __restrict__ x,
                            unsigned short* __restrict__ xb, long long n) {
    long long i = (long long)blockIdx.x * blockDim.x + threadIdx.x;
    long long stride = (long long)gridDim.x * blockDim.x;
    for (; i < n; i += stride) xb[i] = f2bf(x[i]);
}

// Pack W (f32, [K][128]) into WMMA B-fragment order (bf16):
// fragment (wv 0..7, kf 0..K/32-1); lane holds col = wv*16 + (lane&15),
// elements e=0..15 are K = kf*32 + 8*(lane>>4) + e (+8 for e>=8). 32 contiguous B / lane.
__global__ void k_pack_b(const float* __restrict__ W,
                         unsigned short* __restrict__ out, int K) {
    int KF = K >> 5;
    int total = 8 * KF * 32;
    int tid = blockIdx.x * blockDim.x + threadIdx.x;
    if (tid >= total) return;
    int lane = tid & 31;
    int kf   = (tid >> 5) % KF;
    int wv   = tid / (32 * KF);
    int col  = wv * 16 + (lane & 15);
    int kb   = kf * 32 + 8 * (lane >> 4);
    Frag t;
#pragma unroll
    for (int e = 0; e < 16; ++e) t.s[e] = f2bf(W[(size_t)(kb + e + (e >= 8 ? 8 : 0)) * 128 + col]);
    uint4* dst = (uint4*)out + (size_t)((wv * KF + kf) * 32 + lane) * 2;
    dst[0] = t.q[0];
    dst[1] = t.q[1];
}

// --------------------------- CSR binning -----------------------------------

__global__ void k_zero2(int* __restrict__ cnt, int* __restrict__ cursor, int N) {
    int i = blockIdx.x * blockDim.x + threadIdx.x;
    if (i < N) { cnt[i] = 0; cursor[i] = 0; }
}

__global__ void k_count(const long long* __restrict__ ei, int E,
                        int* __restrict__ cnt) {
    int e = blockIdx.x * blockDim.x + threadIdx.x;
    if (e < E) atomicAdd(&cnt[(int)ei[(size_t)E + e]], 1);
}

// Single-block exclusive scan over N counts -> rowptr[0..N]
__global__ __launch_bounds__(1024) void k_scan(const int* __restrict__ cnt,
                                               int* __restrict__ rowptr, int N) {
    __shared__ int buf[1024];
    __shared__ int sCarry;
    if (threadIdx.x == 0) sCarry = 0;
    __syncthreads();
    for (int basei = 0; basei < N; basei += 1024) {
        int i = basei + threadIdx.x;
        int v = (i < N) ? cnt[i] : 0;
        int sum = v;
        buf[threadIdx.x] = sum;
        __syncthreads();
        for (int off = 1; off < 1024; off <<= 1) {
            int t = (threadIdx.x >= off) ? buf[threadIdx.x - off] : 0;
            __syncthreads();
            sum += t;
            buf[threadIdx.x] = sum;
            __syncthreads();
        }
        int carry = sCarry;
        if (i < N) rowptr[i] = carry + sum - v;   // exclusive
        __syncthreads();
        if (threadIdx.x == 1023) sCarry = carry + buf[1023];
        __syncthreads();
    }
    if (threadIdx.x == 0) rowptr[N] = sCarry;
}

// Scatter src node ids into dst-grouped CSR slots
__global__ void k_bin(const long long* __restrict__ ei, int E,
                      const int* __restrict__ rowptr, int* __restrict__ cursor,
                      int* __restrict__ binnedSrc) {
    int e = blockIdx.x * blockDim.x + threadIdx.x;
    if (e < E) {
        int d = (int)ei[(size_t)E + e];
        int slot = atomicAdd(&cursor[d], 1);
        binnedSrc[rowptr[d] + slot] = (int)ei[e];
    }
}

// --------------------------- fused node kernel -----------------------------

#define HSTR 136      // f32/u32 LDS stride per node row (136 % 64 == 8: de-conflicts)
#define ASTR 136      // bf16 A-tile stride (272 B, 16B aligned)
#define GSIZE 32      // edges per group (two 16-row WMMA sub-tiles)
#define FSTRIDE 656   // 640 + 16 pad (bf16)
#define TSTRIDE 144   // 128 + 16 pad (bf16)
#define LNCAP 512     // local-node-id chunk capacity

__global__ __launch_bounds__(256) void k_fused(
    const float* __restrict__ x, const unsigned short* __restrict__ xb,
    const int* __restrict__ rowptr, const int* __restrict__ binnedSrc,
    const unsigned short* __restrict__ WpreP,  const float* __restrict__ b_pre,
    const unsigned short* __restrict__ WpostP, const float* __restrict__ b_post,
    const unsigned short* __restrict__ WlinP,  const float* __restrict__ b_lin,
    float* __restrict__ out, int N, float inv_avg) {
    __shared__ __align__(16) float    sS1[16 * HSTR];
    __shared__ __align__(16) float    sS2[16 * HSTR];
    __shared__ __align__(16) unsigned sMxK[16 * HSTR];
    __shared__ __align__(16) unsigned sMnK[16 * HSTR];
    __shared__ __align__(16) float    sHd[16 * HSTR];
    __shared__ __align__(16) unsigned short sA[GSIZE * ASTR];
    __shared__ __align__(16) unsigned short Ftile[16 * FSTRIDE];
    __shared__ __align__(16) unsigned short Ttile[16 * TSTRIDE];
    __shared__ unsigned char sLn[LNCAP];
    __shared__ int   sRp[17];
    __shared__ float sScale[16];
    __shared__ float sInvD[16];
    __shared__ int   sHas[16];

    const int lane  = threadIdx.x & 31;
    const int wv    = threadIdx.x >> 5;
    const int node0 = blockIdx.x * 16;

    if (threadIdx.x < 17) {
        int n = node0 + threadIdx.x;
        if (n > N) n = N;
        sRp[threadIdx.x] = rowptr[n];
    }
    for (int i = threadIdx.x; i < 16 * HSTR; i += 256) {
        sS1[i] = 0.f;
        sS2[i] = 0.f;
        sMxK[i] = 0u;          // below every valid key
        sMnK[i] = 0xFFFFFFFFu; // above every valid key
    }
    __syncthreads();
    const int base = sRp[0];
    const int T    = sRp[16] - base;

    // W_pre B-fragments resident in registers (K=256 -> 8 frags, 64 VGPRs)
    Frag bpre[8];
#pragma unroll
    for (int kf = 0; kf < 8; ++kf) {
        const uint4* p = (const uint4*)WpreP + (size_t)((wv * 8 + kf) * 32 + lane) * 2;
        bpre[kf].q[0] = p[0];
        bpre[kf].q[1] = p[1];
    }

    const int arow  = lane & 15;
    const int koff  = (lane >> 4) * 8;
    const int gcol  = wv * 16 + (lane & 15);
    const int rbase = (lane >> 4) * 8;

    // ---- hd = x[node] @ W_pre[0:128] + b_pre  (dst half, once per tile) ----
    {
        int row  = threadIdx.x >> 4;
        int col0 = (threadIdx.x & 15) * 8;     // 8 bf16 = 16 B per thread
        int node = node0 + row;
        uint4 a = make_uint4(0, 0, 0, 0);
        if (node < N) a = *(const uint4*)(xb + (size_t)node * 128 + col0);
        *(uint4*)&sA[row * ASTR + col0] = a;
    }
    __syncthreads();
    {
        f32x8 acc = {0.f, 0.f, 0.f, 0.f, 0.f, 0.f, 0.f, 0.f};
#pragma unroll
        for (int kf = 0; kf < 4; ++kf) {
            Frag a;
            a.q[0] = *(const uint4*)&sA[arow * ASTR + kf * 32 + koff];
            a.q[1] = *(const uint4*)&sA[arow * ASTR + kf * 32 + 16 + koff];
            acc = __builtin_amdgcn_wmma_f32_16x16x32_bf16(
                false, a.v, false, bpre[kf].v, (short)0, acc, false, false);
        }
        float bp = b_pre[gcol];
#pragma unroll
        for (int r = 0; r < 8; ++r)
            sHd[(rbase + r) * HSTR + gcol] = acc[r] + bp;
    }
    __syncthreads();

    // Staging geometry for 32-row groups: 8 threads per row, 16 bf16 each
    const int srow  = threadIdx.x >> 3;         // 0..31
    const int scol0 = (threadIdx.x & 7) * 16;   // 0,16,...,112

    // ---- edge loop over this tile's contiguous CSR slice ----
    for (int cb = 0; cb < T; cb += LNCAP) {
        int clen = T - cb;
        if (clen > LNCAP) clen = LNCAP;
        if (threadIdx.x < 16) {          // local-node id per slice position
            int lo = sRp[threadIdx.x] - base;
            int hi = sRp[threadIdx.x + 1] - base;
            if (lo < cb) lo = cb;
            if (hi > cb + clen) hi = cb + clen;
            for (int p = lo; p < hi; ++p) sLn[p - cb] = (unsigned char)threadIdx.x;
        }
        __syncthreads();
        int groups = (clen + GSIZE - 1) >> 5;

        // Prefetch group 0 of this chunk into registers
        uint4 c0 = make_uint4(0, 0, 0, 0), c1 = c0;
        {
            int i = cb + srow;
            if (i < T) {
                int src = binnedSrc[base + i];
                const uint4* p = (const uint4*)(xb + (size_t)src * 128 + scol0);
                c0 = p[0];
                c1 = p[1];
            }
        }
        for (int g = 0; g < groups; ++g) {
            // Commit current group's src rows to LDS
            *(uint4*)&sA[srow * ASTR + scol0]     = c0;
            *(uint4*)&sA[srow * ASTR + scol0 + 8] = c1;
            __syncthreads();

            // Prefetch next group (overlaps with WMMA + LDS reduction below)
            uint4 n0 = make_uint4(0, 0, 0, 0), n1 = n0;
            bool more = (g + 1 < groups);
            if (more) {
                int i = cb + (g + 1) * GSIZE + srow;
                if (i < T) {
                    int src = binnedSrc[base + i];
                    const uint4* p = (const uint4*)(xb + (size_t)src * 128 + scol0);
                    n0 = p[0];
                    n1 = p[1];
                }
            }

            // hs = x[src] @ W_pre[128:256]: two 16-row sub-tiles, 8 WMMAs
            f32x8 acc[2];
            acc[0] = f32x8{0.f, 0.f, 0.f, 0.f, 0.f, 0.f, 0.f, 0.f};
            acc[1] = f32x8{0.f, 0.f, 0.f, 0.f, 0.f, 0.f, 0.f, 0.f};
#pragma unroll
            for (int t = 0; t < 2; ++t) {
#pragma unroll
                for (int kf = 0; kf < 4; ++kf) {
                    Frag a;
                    int rb = (t * 16 + arow) * ASTR + kf * 32 + koff;
                    a.q[0] = *(const uint4*)&sA[rb];
                    a.q[1] = *(const uint4*)&sA[rb + 16];
                    acc[t] = __builtin_amdgcn_wmma_f32_16x16x32_bf16(
                        false, a.v, false, bpre[4 + kf].v, (short)0, acc[t],
                        false, false);
                }
            }

            // LDS reduction: h = hs + hd[ln]; sum / sum^2 / max / min
            const int i0 = cb + g * GSIZE;
#pragma unroll
            for (int t = 0; t < 2; ++t) {
#pragma unroll
                for (int r = 0; r < 8; ++r) {
                    int row = t * 16 + rbase + r;
                    int i = i0 + row;
                    if (i < T) {
                        int ln  = sLn[i - cb];
                        int off = ln * HSTR + gcol;
                        float v = acc[t][r] + sHd[off];
                        atomicAdd(&sS1[off], v);
                        atomicAdd(&sS2[off], v * v);
                        unsigned key = flipf(v);
                        atomicMax(&sMxK[off], key);
                        atomicMin(&sMnK[off], key);
                    }
                }
            }
            __syncthreads();
            c0 = n0;
            c1 = n1;
        }
    }

    // ---- per-node stats ----
    if (threadIdx.x < 16) {
        float d = (float)(sRp[threadIdx.x + 1] - sRp[threadIdx.x]);
        sScale[threadIdx.x] = d * inv_avg;
        sInvD[threadIdx.x]  = 1.f / fmaxf(d, 1.f);
        sHas[threadIdx.x]   = (d > 0.f);
    }
    __syncthreads();

    // ---- build 16x640 bf16 feature tile: [x | mean | max | min | std] ----
#pragma unroll
    for (int sel = 0; sel < 5; ++sel) {
        for (int j = 0; j < 8; ++j) {
            int idx = threadIdx.x + j * 256;  // 0..2047
            int row = idx >> 7;
            int c   = idx & 127;
            int node = node0 + row;
            float v = 0.f;
            if (node < N) {
                int off = row * HSTR + c;
                if (sel == 0) {
                    v = x[(size_t)node * 128 + c];
                } else if (sel == 1) {
                    v = sS1[off] * sInvD[row] * sScale[row];
                } else if (sel == 2) {
                    v = (sHas[row] ? unflipf(sMxK[off]) : 0.f) * sScale[row];
                } else if (sel == 3) {
                    v = (sHas[row] ? unflipf(sMnK[off]) : 0.f) * sScale[row];
                } else {
                    float m   = sS1[off] * sInvD[row];
                    float var = fmaxf(sS2[off] * sInvD[row] - m * m, 0.f);
                    v = sqrtf(var + 1e-5f) * sScale[row];
                }
            }
            Ftile[row * FSTRIDE + sel * 128 + c] = f2bf(v);
        }
    }
    __syncthreads();

    // ---- GEMM1: (16x640) @ W_post + b_post -> bf16 Ttile ----
    f32x8 acc = {0.f, 0.f, 0.f, 0.f, 0.f, 0.f, 0.f, 0.f};
#pragma unroll
    for (int kf = 0; kf < 20; ++kf) {
        Frag b;
        const uint4* bp = (const uint4*)WpostP + (size_t)((wv * 20 + kf) * 32 + lane) * 2;
        b.q[0] = bp[0];
        b.q[1] = bp[1];
        Frag a;
        a.q[0] = *(const uint4*)&Ftile[arow * FSTRIDE + kf * 32 + koff];
        a.q[1] = *(const uint4*)&Ftile[arow * FSTRIDE + kf * 32 + 16 + koff];
        acc = __builtin_amdgcn_wmma_f32_16x16x32_bf16(
            false, a.v, false, b.v, (short)0, acc, false, false);
    }
    {
        float bp = b_post[gcol];
#pragma unroll
        for (int r = 0; r < 8; ++r)
            Ttile[(rbase + r) * TSTRIDE + gcol] = f2bf(acc[r] + bp);
    }
    __syncthreads();

    // ---- GEMM2: (16x128) @ W_lin + b_lin, ReLU ----
    f32x8 acc2 = {0.f, 0.f, 0.f, 0.f, 0.f, 0.f, 0.f, 0.f};
#pragma unroll
    for (int kf = 0; kf < 4; ++kf) {
        Frag b;
        const uint4* bp = (const uint4*)WlinP + (size_t)((wv * 4 + kf) * 32 + lane) * 2;
        b.q[0] = bp[0];
        b.q[1] = bp[1];
        Frag a;
        a.q[0] = *(const uint4*)&Ttile[arow * TSTRIDE + kf * 32 + koff];
        a.q[1] = *(const uint4*)&Ttile[arow * TSTRIDE + kf * 32 + 16 + koff];
        acc2 = __builtin_amdgcn_wmma_f32_16x16x32_bf16(
            false, a.v, false, b.v, (short)0, acc2, false, false);
    }
    {
        float bl = b_lin[gcol];
#pragma unroll
        for (int r = 0; r < 8; ++r) {
            int node = node0 + rbase + r;
            if (node < N)
                out[(size_t)node * 128 + gcol] = fmaxf(acc2[r] + bl, 0.f);
        }
    }
}

// --------------------------- launch ----------------------------------------

extern "C" void kernel_launch(void* const* d_in, const int* in_sizes, int n_in,
                              void* d_out, int out_size, void* d_ws, size_t ws_size,
                              hipStream_t stream) {
    const float* x      = (const float*)d_in[0];
    const float* W_pre  = (const float*)d_in[1];
    const float* b_pre  = (const float*)d_in[2];
    const float* W_post = (const float*)d_in[3];
    const float* b_post = (const float*)d_in[4];
    const float* W_lin  = (const float*)d_in[5];
    const float* b_lin  = (const float*)d_in[6];
    const long long* ei = (const long long*)d_in[7];

    const int N = in_sizes[0] / 128;
    const int E = in_sizes[7] / 2;
    const float inv_avg = (float)N / (float)E;  // 1/mean(deg); mean = E/N exactly

    char* ws = (char*)d_ws;
    size_t off = 0;
    auto carve = [&](size_t bytes) {
        char* p = ws + off;
        off = (off + bytes + 255) & ~(size_t)255;
        return p;
    };
    unsigned short* xb     = (unsigned short*)carve((size_t)N * 128 * 2);
    unsigned short* WpreP  = (unsigned short*)carve(256 * 128 * 2);
    unsigned short* WpostP = (unsigned short*)carve(640 * 128 * 2);
    unsigned short* WlinP  = (unsigned short*)carve(128 * 128 * 2);
    int* cnt       = (int*)carve((size_t)N * 4);
    int* cursor    = (int*)carve((size_t)N * 4);
    int* rowptr    = (int*)carve(((size_t)N + 1) * 4);
    int* binnedSrc = (int*)carve((size_t)E * 4);
    (void)ws_size;

    const long long nFeat = (long long)N * 128;

    k_convert_x<<<2048, 256, 0, stream>>>(x, xb, nFeat);
    k_pack_b<<<(8 * 8 * 32 + 255) / 256, 256, 0, stream>>>(W_pre, WpreP, 256);
    k_pack_b<<<(8 * 20 * 32 + 255) / 256, 256, 0, stream>>>(W_post, WpostP, 640);
    k_pack_b<<<(8 * 4 * 32 + 255) / 256, 256, 0, stream>>>(W_lin, WlinP, 128);

    k_zero2<<<(N + 255) / 256, 256, 0, stream>>>(cnt, cursor, N);
    k_count<<<(E + 255) / 256, 256, 0, stream>>>(ei, E, cnt);
    k_scan<<<1, 1024, 0, stream>>>(cnt, rowptr, N);
    k_bin<<<(E + 255) / 256, 256, 0, stream>>>(ei, E, rowptr, cursor, binnedSrc);

    k_fused<<<(N + 15) / 16, 256, 0, stream>>>(x, xb, rowptr, binnedSrc,
                                               WpreP, b_pre, WpostP, b_post,
                                               WlinP, b_lin,
                                               (float*)d_out, N, inv_avg);
}
